// DeformableNeighborhoodAttention_3015067042135
// MI455X (gfx1250) — matrix operand order repaired and checked by reading, not compiled, not run on hardware
//
#include <hip/hip_runtime.h>
#include <hip/hip_bf16.h>
#include <math.h>

// DeformableNeighborhoodAttention for MI455X (gfx1250, wave32).
// conv1x1 GEMMs -> v_wmma_f32_16x16x32_f16 with multi-N-tile accumulation so
// the activation matrix is streamed from HBM ~once; offsets/sampling/attention
// remain VALU (irregular per-pixel windows are not WMMA-shaped).

typedef __attribute__((ext_vector_type(16))) _Float16 v16h;
typedef __attribute__((ext_vector_type(8)))  float    v8f;

constexpr int Bn = 4, Cn = 128, Hn = 64, Wn = 64, Gn = 4, HDn = 32, Kn = 7;
constexpr int HWn = Hn * Wn;
constexpr float EPSf   = 1e-6f;
constexpr float SCALEf = 0.17677669529663687f;   // HD^-0.5 = 1/sqrt(32)

__device__ __forceinline__ float wsum(float v) {
  v += __shfl_xor(v, 16, 32);
  v += __shfl_xor(v, 8, 32);
  v += __shfl_xor(v, 4, 32);
  v += __shfl_xor(v, 2, 32);
  v += __shfl_xor(v, 1, 32);
  return v;
}
__device__ __forceinline__ float wmaxr(float v) {
  v = fmaxf(v, __shfl_xor(v, 16, 32));
  v = fmaxf(v, __shfl_xor(v, 8, 32));
  v = fmaxf(v, __shfl_xor(v, 4, 32));
  v = fmaxf(v, __shfl_xor(v, 2, 32));
  v = fmaxf(v, __shfl_xor(v, 1, 32));
  return v;
}

// ---------------------------------------------------------------------------
// Fragment layouts used here (wave32), 16x16x32 f16->f32; A/B k order is contiguous, not the ISA table order:
//   A 16x32 : lane -> M=lane&15 ; elem j -> K = 16h + j, h=lane>>4
//   B 32x16 : lane -> N=lane&15 ; elem j -> K = 16h + j
//   D 16x16 : lane -> N=lane&15 ; VGPR r -> M = 8h + r
// ---------------------------------------------------------------------------
__device__ __forceinline__ v16h load_a_frag(const float* __restrict__ inb,
                                            int k0, int half) {
  v16h a;
#pragma unroll
  for (int j = 0; j < 16; ++j) {
    const int koff = 16 * half + j;
    a[j] = (_Float16)inb[(size_t)(k0 + koff) * HWn];
  }
  return a;
}
__device__ __forceinline__ v16h load_b_frag(const float* __restrict__ Wt,
                                            int n0, int k0, int half, int m) {
  const float* wr = Wt + (size_t)(n0 + m) * Cn + k0 + 16 * half;
  v16h bf;
#pragma unroll
  for (int j = 0; j < 16; ++j) bf[j] = (_Float16)wr[j];
  return bf;
}

// conv1x1 as GEMM: out[b,n,p] = sum_k in[b,k,p] * Wt[n,k] + bias[n]
// One wave -> 16 pixels x ALL 128 output channels (8 accumulators); the A
// fragment is loaded once per K-step and reused for 8 WMMAs.
__global__ __launch_bounds__(256) void conv1x1_wmma_kernel(
    const float* __restrict__ in, const float* __restrict__ Wt,
    const float* __restrict__ bias, float* __restrict__ out)
{
  const int lane  = threadIdx.x & 31;
  const int mtile = (int)((blockIdx.x * blockDim.x + threadIdx.x) >> 5); // 0..1023
  const int b     = mtile >> 8;            // HW/16 = 256 M tiles per batch
  const int p0    = (mtile & 255) * 16;
  const int half  = lane >> 4;
  const int m     = lane & 15;

  const float* inb = in + (size_t)b * Cn * HWn + p0 + m;
  v8f acc[8] = {};
  for (int k0 = 0; k0 < Cn; k0 += 32) {
    const v16h a = load_a_frag(inb, k0, half);
#pragma unroll
    for (int nt = 0; nt < 8; ++nt) {
      const v16h bf = load_b_frag(Wt, nt * 16, k0, half, m);
      acc[nt] = __builtin_amdgcn_wmma_f32_16x16x32_f16(false, a, false, bf,
                                                       (short)0, acc[nt], false, false);
    }
  }
#pragma unroll
  for (int nt = 0; nt < 8; ++nt) {
    const int n = nt * 16 + m;
    const float bn = bias[n];
    float* op = out + ((size_t)b * Cn + n) * HWn + p0 + 8 * half;
#pragma unroll
    for (int r = 0; r < 8; ++r) op[r] = acc[nt][r] + bn;
  }
}

// Fused k = xs*Wk+bk and v = xs*Wv+bv. One wave -> 16 pixels x 64 output
// channels for BOTH k and v (4 N-tiles x 2 outputs = 8 accumulators); the A
// fragment is reused for 8 WMMAs per K-step, A re-read only 2x total.
__global__ __launch_bounds__(256) void conv1x1_kv_wmma_kernel(
    const float* __restrict__ in,
    const float* __restrict__ Wk, const float* __restrict__ bk,
    const float* __restrict__ Wv, const float* __restrict__ bv,
    float* __restrict__ kout, float* __restrict__ vout)
{
  const int lane  = threadIdx.x & 31;
  const int wid   = (int)((blockIdx.x * blockDim.x + threadIdx.x) >> 5); // 0..2047
  const int ng    = wid & 1;               // N half: channels [ng*64, ng*64+64)
  const int mtile = wid >> 1;              // 0..1023
  const int b     = mtile >> 8;
  const int p0    = (mtile & 255) * 16;
  const int half  = lane >> 4;
  const int m     = lane & 15;

  const float* inb = in + (size_t)b * Cn * HWn + p0 + m;
  v8f ak[4] = {}, av[4] = {};
  for (int k0 = 0; k0 < Cn; k0 += 32) {
    const v16h a = load_a_frag(inb, k0, half);
#pragma unroll
    for (int nt = 0; nt < 4; ++nt) {
      const int n0 = ng * 64 + nt * 16;
      const v16h bkf = load_b_frag(Wk, n0, k0, half, m);
      ak[nt] = __builtin_amdgcn_wmma_f32_16x16x32_f16(false, a, false, bkf,
                                                      (short)0, ak[nt], false, false);
      const v16h bvf = load_b_frag(Wv, n0, k0, half, m);
      av[nt] = __builtin_amdgcn_wmma_f32_16x16x32_f16(false, a, false, bvf,
                                                      (short)0, av[nt], false, false);
    }
  }
#pragma unroll
  for (int nt = 0; nt < 4; ++nt) {
    const int n = ng * 64 + nt * 16 + m;
    const float bkn = bk[n];
    const float bvn = bv[n];
    float* kp = kout + ((size_t)b * Cn + n) * HWn + p0 + 8 * half;
    float* vp = vout + ((size_t)b * Cn + n) * HWn + p0 + 8 * half;
#pragma unroll
    for (int r = 0; r < 8; ++r) { kp[r] = ak[nt][r] + bkn; vp[r] = av[nt][r] + bvn; }
  }
}

// ---------------------------------------------------------------------------
// Offset branch + grid sample. One wave per (bg,h,w) pixel, lane = channel
// (HD == 32 == wave32). Control flow is lane-uniform -> shuffles are safe.
// ---------------------------------------------------------------------------
__global__ __launch_bounds__(256) void offset_sample_kernel(
    const float* __restrict__ q, const float* __restrict__ x,
    const float* __restrict__ dww, const float* __restrict__ dwb,
    const float* __restrict__ lnw, const float* __restrict__ lnb,
    const float* __restrict__ pww, float* __restrict__ xs)
{
  const int lane = threadIdx.x & 31;                        // channel in group
  const int wid  = (int)((blockIdx.x * blockDim.x + threadIdx.x) >> 5);
  const int w  = wid & (Wn - 1);
  const int h  = (wid >> 6) & (Hn - 1);
  const int bg = wid >> 12;                                 // 0..15 (B*G)
  const size_t cbase = ((size_t)bg * HDn + lane) * HWn;     // == (b*C+gc)*HW

  // 7x7 depthwise conv on q (zero pad 3)
  float o = dwb[lane];
#pragma unroll
  for (int ky = 0; ky < Kn; ++ky) {
    const int yy = h + ky - 3;
    if (yy < 0 || yy >= Hn) continue;
#pragma unroll
    for (int kx = 0; kx < Kn; ++kx) {
      const int xx = w + kx - 3;
      if (xx < 0 || xx >= Wn) continue;
      o += q[cbase + yy * Wn + xx] * dww[lane * (Kn * Kn) + ky * Kn + kx];
    }
  }
  // layernorm over the 32 channels of the group (wave reduction)
  const float mu  = wsum(o) * (1.f / 32.f);
  const float d   = o - mu;
  const float var = wsum(d * d) * (1.f / 32.f);
  float a = lnw[lane] * (d * rsqrtf(var + EPSf)) + lnb[lane];
  // exact gelu
  a = 0.5f * a * (1.f + erff(a * 0.70710678118654752f));
  // pointwise HD -> 2 (wave reductions), tanh, scale
  const float off0 = wsum(a * pww[lane]);          // y channel
  const float off1 = wsum(a * pww[HDn + lane]);    // x channel
  const float offy = tanhf(off0) * (1.f / (float)(Hn - 1));
  const float offx = tanhf(off1) * (1.f / (float)(Wn - 1));
  const float refy = (0.5f + (float)h) * (2.f / (float)(Hn - 1)) - 1.f;
  const float refx = (0.5f + (float)w) * (2.f / (float)(Wn - 1)) - 1.f;
  const float gy = (refy + offy + 1.f) * 0.5f * (float)(Hn - 1);
  const float gx = (refx + offx + 1.f) * 0.5f * (float)(Wn - 1);
  const float y0f = floorf(gy), x0f = floorf(gx);

  float acc = 0.f;
#pragma unroll
  for (int dy = 0; dy < 2; ++dy) {
#pragma unroll
    for (int dx = 0; dx < 2; ++dx) {
      const float yi = y0f + (float)dy, xi = x0f + (float)dx;
      const float wgt = (1.f - fabsf(gx - xi)) * (1.f - fabsf(gy - yi));
      const bool valid = (xi >= 0.f) && (xi <= (float)(Wn - 1)) &&
                         (yi >= 0.f) && (yi <= (float)(Hn - 1));
      const int yc = (int)fminf(fmaxf(yi, 0.f), (float)(Hn - 1));
      const int xc = (int)fminf(fmaxf(xi, 0.f), (float)(Wn - 1));
      acc += x[cbase + yc * Wn + xc] * (valid ? wgt : 0.f);
    }
  }
  xs[cbase + h * Wn + w] = acc;
}

// ---------------------------------------------------------------------------
// Neighborhood attention (49-key clipped window) + fused lepe (3x3 dw-conv).
// One wave per query (b,g,h,w); lane = head channel. Logits/softmax weights
// staged in a private 64-float LDS row per wave.
// ---------------------------------------------------------------------------
__global__ __launch_bounds__(256) void attn_kernel(
    const float* __restrict__ q, const float* __restrict__ kbuf,
    const float* __restrict__ vbuf, const float* __restrict__ rpew,
    const float* __restrict__ rpeb, float* __restrict__ att)
{
  __shared__ float slog[8][64];
  const int lane = threadIdx.x & 31;
  const int wv   = threadIdx.x >> 5;
  const int wid  = (int)(blockIdx.x * 8 + wv);
  const int w  = wid & (Wn - 1);
  const int h  = (wid >> 6) & (Hn - 1);
  const int bg = wid >> 12;
  const size_t cbase = ((size_t)bg * HDn + lane) * HWn;     // == (b*C+gc)*HW

  const float qs = q[cbase + h * Wn + w] * SCALEf;
  const int r0 = min(max(h - 3, 0), Hn - Kn);
  const int c0 = min(max(w - 3, 0), Wn - Kn);

  for (int p = 0; p < 49; ++p) {
    const int ky = r0 + p / 7, kx = c0 + p % 7;
    const float s = wsum(qs * kbuf[cbase + ky * Wn + kx]);
    if (lane == 0) slog[wv][p] = s;
  }
  const float l1 = slog[wv][lane];
  const float l2 = (lane < 17) ? slog[wv][lane + 32] : -3.4e38f;
  const float mx = wmaxr(fmaxf(l1, l2));
  const float e1 = expf(l1 - mx);
  const float e2 = (lane < 17) ? expf(l2 - mx) : 0.f;
  const float inv = 1.f / wsum(e1 + e2);
  slog[wv][lane] = e1 * inv;
  if (lane < 17) slog[wv][lane + 32] = e2 * inv;

  float acc = 0.f;
  for (int p = 0; p < 49; ++p) {
    const int ky = r0 + p / 7, kx = c0 + p % 7;
    acc += slog[wv][p] * vbuf[cbase + ky * Wn + kx];
  }

  // lepe: 3x3 depthwise conv of q with zero pad 1, per global channel gc
  const int gc = (bg & (Gn - 1)) * HDn + lane;
  float lepe = rpeb[gc];
#pragma unroll
  for (int dy = 0; dy < 3; ++dy) {
    const int yy = h + dy - 1;
    if (yy < 0 || yy >= Hn) continue;
#pragma unroll
    for (int dx = 0; dx < 3; ++dx) {
      const int xx = w + dx - 1;
      if (xx < 0 || xx >= Wn) continue;
      lepe += q[cbase + yy * Wn + xx] * rpew[gc * 9 + dy * 3 + dx];
    }
  }
  att[cbase + h * Wn + w] = acc + lepe;
}

// ---------------------------------------------------------------------------
extern "C" void kernel_launch(void* const* d_in, const int* in_sizes, int n_in,
                              void* d_out, int out_size, void* d_ws, size_t ws_size,
                              hipStream_t stream) {
  (void)in_sizes; (void)n_in; (void)out_size; (void)ws_size;
  const float* x    = (const float*)d_in[0];
  const float* Wq   = (const float*)d_in[1];
  const float* bq   = (const float*)d_in[2];
  const float* Wk   = (const float*)d_in[3];
  const float* bk   = (const float*)d_in[4];
  const float* Wv   = (const float*)d_in[5];
  const float* bv   = (const float*)d_in[6];
  const float* Wo   = (const float*)d_in[7];
  const float* bo   = (const float*)d_in[8];
  const float* dww  = (const float*)d_in[9];
  const float* dwb  = (const float*)d_in[10];
  const float* lnw  = (const float*)d_in[11];
  const float* lnb  = (const float*)d_in[12];
  const float* pww  = (const float*)d_in[13];
  const float* rpew = (const float*)d_in[14];
  const float* rpeb = (const float*)d_in[15];
  float* out = (float*)d_out;

  const size_t NF = (size_t)Bn * Cn * HWn;   // 2M floats = 8 MB per tensor
  float* qb   = (float*)d_ws;                // [0,  8MB)
  float* xsb  = qb + NF;                     // [8, 16MB)
  float* kb   = xsb + NF;                    // [16,24MB)
  float* vb   = kb + NF;                     // [24,32MB)
  float* attb = xsb;                         // xs is dead after the k/v GEMM

  const dim3 blk(256);
  // q/o GEMM: 1024 waves (16px x 128ch each) -> 128 blocks
  // kv GEMM : 2048 waves (16px x 64ch x {k,v})-> 256 blocks
  // pixel kernels: 65536 waves -> 8192 blocks
  conv1x1_wmma_kernel   <<<128,  blk, 0, stream>>>(x,   Wq, bq, qb);
  offset_sample_kernel  <<<8192, blk, 0, stream>>>(qb, x, dww, dwb, lnw, lnb, pww, xsb);
  conv1x1_kv_wmma_kernel<<<256,  blk, 0, stream>>>(xsb, Wk, bk, Wv, bv, kb, vb);
  attn_kernel           <<<8192, blk, 0, stream>>>(qb, kb, vb, rpew, rpeb, attb);
  conv1x1_wmma_kernel   <<<128,  blk, 0, stream>>>(attb, Wo, bo, out);
}
